// InputBlock_22531398435247
// MI455X (gfx1250) — matrix operand order repaired and checked by reading, not compiled, not run on hardware
//
#include <hip/hip_runtime.h>
#include <hip/hip_bf16.h>
#include <stdint.h>

// ---------------------------------------------------------------------------
// Problem constants (from the reference)
// ---------------------------------------------------------------------------
#define NROWS 4096
#define NBLK  4
#define DIN   4000
#define DE    128
#define DB    8
#define DLAT  512
#define KDIM  (DIN + DE + DB)   // 4136
#define KPAD  4160              // padded to a multiple of 32 (130 chunks)
#define KCH   32                // WMMA bf16 K per instruction
#define NCH   (KPAD / KCH)      // 130
#define ALPHA 0.3f

#define TILE_M 128
#define TILE_N 128
#define LDS_STRIDE 80           // bytes per 32-half row (64B data + 16B pad)

// workspace layout
#define WS_A_BYTES ((size_t)NBLK * NROWS * KPAD * 2)   // 136,314,880 B
#define WS_W_BYTES ((size_t)NBLK * DLAT  * KPAD * 2)   //  17,039,360 B

typedef __attribute__((ext_vector_type(16))) __bf16 bf16x16;
typedef __attribute__((ext_vector_type(8)))  __bf16 bf16x8;
typedef __attribute__((ext_vector_type(8)))  float  f32x8;
typedef __attribute__((ext_vector_type(4)))  unsigned int u32x4;

// ---------------------------------------------------------------------------
// Phase 1a: build bf16 A = [mask?(x-mean):0 | embed | batches | 0-pad]
// Layout: A[b][n][KPAD], K contiguous. One thread = 8 consecutive K (16 B out).
// All segment boundaries (4000, 4128, 4136) are multiples of 8.
// ---------------------------------------------------------------------------
__global__ __launch_bounds__(256) void pack_a_kernel(
    const float* __restrict__ x, const unsigned char* __restrict__ mask,
    const float* __restrict__ embed, const float* __restrict__ batches,
    const float* __restrict__ mean_vals, __bf16* __restrict__ A)
{
    const int G = KPAD / 8;  // 520 groups per (b, n)
    size_t idx = (size_t)blockIdx.x * blockDim.x + threadIdx.x;
    size_t total = (size_t)NBLK * NROWS * G;
    if (idx >= total) return;
    int g = (int)(idx % G);
    int n = (int)((idx / G) % NROWS);
    int b = (int)(idx / ((size_t)G * NROWS));
    int k0 = g * 8;

    float v[8];
    if (g < DIN / 8) {                          // masked, mean-centered x
        size_t xoff = (size_t)n * (NBLK * DIN) + (size_t)b * DIN + k0;
        const float4* x4 = (const float4*)(x + xoff);
        float4 lo = x4[0], hi = x4[1];
        const float4* m4 = (const float4*)(mean_vals + (size_t)b * DIN + k0);
        float4 ml = m4[0], mh = m4[1];
        uint2 mk = *(const uint2*)(mask + xoff); // 8 bool bytes
        unsigned long long mb = ((unsigned long long)mk.y << 32) | mk.x;
        float t[8] = {lo.x - ml.x, lo.y - ml.y, lo.z - ml.z, lo.w - ml.w,
                      hi.x - mh.x, hi.y - mh.y, hi.z - mh.z, hi.w - mh.w};
#pragma unroll
        for (int j = 0; j < 8; ++j)
            v[j] = ((mb >> (8 * j)) & 0xffull) ? t[j] : 0.0f;
    } else if (g < (DIN + DE) / 8) {            // embed
        size_t eoff = (size_t)n * (NBLK * DE) + (size_t)b * DE + (k0 - DIN);
        const float4* e4 = (const float4*)(embed + eoff);
        float4 lo = e4[0], hi = e4[1];
        v[0]=lo.x; v[1]=lo.y; v[2]=lo.z; v[3]=lo.w;
        v[4]=hi.x; v[5]=hi.y; v[6]=hi.z; v[7]=hi.w;
    } else if (g == (DIN + DE) / 8) {           // batches (exactly 8 wide)
        const float4* b4 = (const float4*)(batches + (size_t)n * DB);
        float4 lo = b4[0], hi = b4[1];
        v[0]=lo.x; v[1]=lo.y; v[2]=lo.z; v[3]=lo.w;
        v[4]=hi.x; v[5]=hi.y; v[6]=hi.z; v[7]=hi.w;
    } else {                                    // K padding
#pragma unroll
        for (int j = 0; j < 8; ++j) v[j] = 0.0f;
    }

    bf16x8 o;
#pragma unroll
    for (int j = 0; j < 8; ++j) o[j] = (__bf16)v[j];
    *(bf16x8*)(A + ((size_t)(b * NROWS + n)) * KPAD + k0) = o;
}

// ---------------------------------------------------------------------------
// Phase 1b: W[b][k][o] fp32  ->  Wp[b][o][KPAD] bf16 (K contiguous per column)
// Thread index has `o` fastest so global reads are coalesced per K.
// ---------------------------------------------------------------------------
__global__ __launch_bounds__(256) void pack_w_kernel(
    const float* __restrict__ W, __bf16* __restrict__ Wp)
{
    const int G = KPAD / 8;  // 520
    size_t idx = (size_t)blockIdx.x * blockDim.x + threadIdx.x;
    size_t total = (size_t)NBLK * DLAT * G;
    if (idx >= total) return;
    int o = (int)(idx % DLAT);
    int g = (int)((idx / DLAT) % G);
    int b = (int)(idx / ((size_t)DLAT * G));
    int k0 = g * 8;

    bf16x8 ov;
#pragma unroll
    for (int j = 0; j < 8; ++j) {
        int k = k0 + j;
        float v = (k < KDIM) ? W[((size_t)b * KDIM + k) * DLAT + o] : 0.0f;
        ov[j] = (__bf16)v;
    }
    *(bf16x8*)(Wp + ((size_t)(b * DLAT + o)) * KPAD + k0) = ov;
}

// ---------------------------------------------------------------------------
// One K-chunk of compute for one wave: A fragment + 8 pipelined B fragments.
// B fragments are register-double-buffered so ds_load latency overlaps WMMA.
// ---------------------------------------------------------------------------
__device__ __forceinline__ void compute_chunk(
    const unsigned char* bufA, const unsigned char* bufB,
    int w, int l16, int lh, f32x8 acc[8])
{
    union F { u32x4 u[2]; bf16x16 v; };
    F fa;
    const unsigned char* rA = bufA + (w * 16 + l16) * LDS_STRIDE + lh * 32;
    fa.u[0] = *(const u32x4*)(rA);        // K 0-7  (lanes<16) / 16-23
    fa.u[1] = *(const u32x4*)(rA + 16);   // K 8-15 (lanes<16) / 24-31

    F fb[2];
    const unsigned char* rB0 = bufB + l16 * LDS_STRIDE + lh * 32;
    fb[0].u[0] = *(const u32x4*)(rB0);
    fb[0].u[1] = *(const u32x4*)(rB0 + 16);
#pragma unroll
    for (int j = 0; j < 8; ++j) {
        if (j + 1 < 8) {   // issue next fragment's loads before this WMMA
            const unsigned char* rB =
                bufB + ((j + 1) * 16 + l16) * LDS_STRIDE + lh * 32;
            fb[(j + 1) & 1].u[0] = *(const u32x4*)(rB);
            fb[(j + 1) & 1].u[1] = *(const u32x4*)(rB + 16);
        }
        acc[j] = __builtin_amdgcn_wmma_f32_16x16x32_bf16(
            false, fa.v, false, fb[j & 1].v, (short)0, acc[j], false, false);
    }
}

// ---------------------------------------------------------------------------
// Phase 2: grouped GEMM, 128x128 tile / 256 threads (8 wave32 waves).
// Double-buffered LDS: globals for chunk c+1 are issued before computing
// chunk c, so HBM/L2 latency hides behind 8 WMMAs + LDS traffic.
// ---------------------------------------------------------------------------
__global__ __launch_bounds__(256) void grouped_gemm_kernel(
    const __bf16* __restrict__ A, const __bf16* __restrict__ Wp,
    const float* __restrict__ bias, float* __restrict__ out)
{
    __shared__ __align__(16) unsigned char lA[2][TILE_M * LDS_STRIDE];
    __shared__ __align__(16) unsigned char lB[2][TILE_N * LDS_STRIDE];

    const int b    = blockIdx.z;
    const int m0   = blockIdx.x * TILE_M;
    const int n0   = blockIdx.y * TILE_N;
    const int t    = threadIdx.x;
    const int w    = t >> 5;        // wave 0..7
    const int lane = t & 31;
    const int lh   = lane >> 4;     // lane half: 0 or 1
    const int l16  = lane & 15;

    // staging: thread t loads 32B half `sh` of 64B row `sr` of each tile
    const int sr = t >> 1;
    const int sh = t & 1;
    const __bf16* Ag = A  + ((size_t)(b * NROWS + m0 + sr)) * KPAD + sh * 16;
    const __bf16* Bg = Wp + ((size_t)(b * DLAT  + n0 + sr)) * KPAD + sh * 16;
    const int soff = sr * LDS_STRIDE + sh * 32;

    // accumulators seeded with bias (each lane's C column is n0+j*16+l16)
    f32x8 acc[8];
#pragma unroll
    for (int j = 0; j < 8; ++j) {
        float bv = bias[b * DLAT + n0 + j * 16 + l16];
#pragma unroll
        for (int r = 0; r < 8; ++r) acc[j][r] = bv;
    }

    // ---- prologue: stage chunk 0 into buffer 0 ----
    {
        u32x4 a0 = *(const u32x4*)(Ag);
        u32x4 a1 = *(const u32x4*)(Ag + 8);
        u32x4 b0 = *(const u32x4*)(Bg);
        u32x4 b1 = *(const u32x4*)(Bg + 8);
        Ag += KCH; Bg += KCH;
        *(u32x4*)(lA[0] + soff)      = a0;
        *(u32x4*)(lA[0] + soff + 16) = a1;
        *(u32x4*)(lB[0] + soff)      = b0;
        *(u32x4*)(lB[0] + soff + 16) = b1;
        __syncthreads();
    }

    // ---- main loop: load chunk c+1, compute chunk c, stage c+1, barrier ----
    for (int c = 0; c < NCH - 1; ++c) {
        const int cur = c & 1, nxt = cur ^ 1;
        u32x4 a0 = *(const u32x4*)(Ag);
        u32x4 a1 = *(const u32x4*)(Ag + 8);
        u32x4 b0 = *(const u32x4*)(Bg);
        u32x4 b1 = *(const u32x4*)(Bg + 8);
        Ag += KCH; Bg += KCH;
        __builtin_prefetch((const void*)(Ag + KCH), 0, 3);  // warm +2 chunk
        __builtin_prefetch((const void*)(Bg + KCH), 0, 3);

        compute_chunk(lA[cur], lB[cur], w, l16, lh, acc);   // hides load latency

        *(u32x4*)(lA[nxt] + soff)      = a0;   // waits loadcnt here, after WMMAs
        *(u32x4*)(lA[nxt] + soff + 16) = a1;
        *(u32x4*)(lB[nxt] + soff)      = b0;
        *(u32x4*)(lB[nxt] + soff + 16) = b1;
        __syncthreads();
    }

    // ---- tail chunk ----
    compute_chunk(lA[(NCH - 1) & 1], lB[(NCH - 1) & 1], w, l16, lh, acc);

    // epilogue: LeakyReLU + store. C layout: VGPR r -> row r (lanes<16) / 8+r.
    const int rowbase = m0 + w * 16 + lh * 8;
#pragma unroll
    for (int j = 0; j < 8; ++j) {
        int col = b * DLAT + n0 + j * 16 + l16;
#pragma unroll
        for (int r = 0; r < 8; ++r) {
            float h = acc[j][r];
            h = (h >= 0.0f) ? h : ALPHA * h;
            out[(size_t)(rowbase + r) * (NBLK * DLAT) + col] = h;
        }
    }
}

// ---------------------------------------------------------------------------
extern "C" void kernel_launch(void* const* d_in, const int* in_sizes, int n_in,
                              void* d_out, int out_size, void* d_ws, size_t ws_size,
                              hipStream_t stream)
{
    const float*         x       = (const float*)d_in[0];
    const unsigned char* mask    = (const unsigned char*)d_in[1];
    const float*         embed   = (const float*)d_in[2];
    const float*         batches = (const float*)d_in[3];
    const float*         meanv   = (const float*)d_in[4];
    const float*         W       = (const float*)d_in[5];
    const float*         bias    = (const float*)d_in[6];
    float*               out     = (float*)d_out;

    __bf16* Abf = (__bf16*)d_ws;
    __bf16* Wbf = (__bf16*)((char*)d_ws + WS_A_BYTES);

    {
        size_t total = (size_t)NBLK * NROWS * (KPAD / 8);
        int blocks = (int)((total + 255) / 256);
        pack_a_kernel<<<blocks, 256, 0, stream>>>(x, mask, embed, batches,
                                                  meanv, Abf);
    }
    {
        size_t total = (size_t)NBLK * DLAT * (KPAD / 8);
        int blocks = (int)((total + 255) / 256);
        pack_w_kernel<<<blocks, 256, 0, stream>>>(W, Wbf);
    }
    {
        dim3 grid(NROWS / TILE_M, DLAT / TILE_N, NBLK);
        grouped_gemm_kernel<<<grid, 256, 0, stream>>>(Abf, Wbf, bias, out);
    }
}
